// InS_mechanism_47923245089245
// MI455X (gfx1250) — matrix operand order repaired and checked by reading, not compiled, not run on hardware
//
#include <hip/hip_runtime.h>
#include <math.h>

// ---------------------------------------------------------------------------
// Types for CDNA5 WMMA (wave32)
// ---------------------------------------------------------------------------
typedef __attribute__((ext_vector_type(16))) __bf16        v16bf;
typedef __attribute__((ext_vector_type(8)))  unsigned int  v8u;
typedef __attribute__((ext_vector_type(8)))  float         v8f;

#define EPSV 1e-8f

// One 8-element K-group of a packed operand row: 8x bf16 hi then 8x bf16 lo.
// Same bytes/element as fp32 (4B), but zero conversion ALU at GEMM time.
struct bgroup { uint4 hi; uint4 lo; };   // 32 bytes

__device__ __forceinline__ unsigned short f32_to_bf16_rne(float f) {
    unsigned int u = __float_as_uint(f);
    unsigned int r = u + 0x7FFFu + ((u >> 16) & 1u);   // round-to-nearest-even
    return (unsigned short)(r >> 16);
}
__device__ __forceinline__ float bf16_bits_to_f32(unsigned short h) {
    return __uint_as_float(((unsigned int)h) << 16);
}

// Split 8 floats -> hi/lo bf16 bit-packs (bf16x3 fp32-emulation, err ~2^-16).
__device__ __forceinline__ void split8(const float* __restrict__ xs,
                                       uint4& hv, uint4& lv) {
    unsigned int h[8], l[8];
#pragma unroll
    for (int i = 0; i < 8; ++i) {
        unsigned short hb = f32_to_bf16_rne(xs[i]);
        float rem = xs[i] - bf16_bits_to_f32(hb);
        h[i] = hb;
        l[i] = f32_to_bf16_rne(rem);
    }
    hv.x = h[0] | (h[1] << 16); hv.y = h[2] | (h[3] << 16);
    hv.z = h[4] | (h[5] << 16); hv.w = h[6] | (h[7] << 16);
    lv.x = l[0] | (l[1] << 16); lv.y = l[2] | (l[3] << 16);
    lv.z = l[4] | (l[5] << 16); lv.w = l[6] | (l[7] << 16);
}

// Assemble a 16-element bf16 WMMA operand from two 8-element bit groups.
__device__ __forceinline__ v16bf comb(uint4 a, uint4 b) {
    v8u w;
    w[0] = a.x; w[1] = a.y; w[2] = a.z; w[3] = a.w;
    w[4] = b.x; w[5] = b.y; w[6] = b.z; w[7] = b.w;
    return __builtin_bit_cast(v16bf, w);
}

__device__ __forceinline__ v8f wmma_bf16(v16bf a, v16bf b, v8f c) {
    // D(16x16,f32) = A(16x32,bf16) x B(32x16,bf16) + C
    return __builtin_amdgcn_wmma_f32_16x16x32_bf16(
        false, a, false, b, (short)0, c, false, false);
}

// ---------------------------------------------------------------------------
// Per-wave 16x64 GEMM tile on packed operands: acc[t] += Arow(16xK)*Bcols(Kx16)
//   Arow : packed row of this lane's M-row, Kgrp groups of 8 K-elements
//   Bsrc : packed rows, row n supplies COLUMN n of B; row stride = Kgrp groups
// A lane layout (ISA 7.12.2): lane<16 K={s..s+7, s+16..s+23}, lane>=16 +8.
// B lane layout: column = lane&15, K same as A: {s..s+7, s+16..s+23}, lane>=16 +8.
// ---------------------------------------------------------------------------
__device__ __forceinline__ void gemm_tile_16x64_pk(const bgroup* __restrict__ Arow,
                                                   const bgroup* __restrict__ Bsrc,
                                                   int ncol0, int Kgrp, int lane,
                                                   v8f acc[4]) {
    const int ga = (lane < 16) ? 0 : 1;   // A K-chunk select (aoff/8)
    const int gb = (lane < 16) ? 0 : 1;   // B K-chunk select (same as A)
    const int ln = lane & 15;
    const bgroup* Brow0 = Bsrc + (size_t)(ncol0 + ln) * (size_t)Kgrp;

    for (int gs = 0; gs < Kgrp; gs += 4) {           // 4 groups = 32 K / step
        bgroup a0 = Arow[gs + ga];
        bgroup a1 = Arow[gs + 2 + ga];
        v16bf ahi = comb(a0.hi, a1.hi);
        v16bf alo = comb(a0.lo, a1.lo);
#pragma unroll
        for (int t = 0; t < 4; ++t) {
            const bgroup* Brow = Brow0 + (size_t)(t * 16) * (size_t)Kgrp;
            bgroup b0 = Brow[gs + gb];
            bgroup b1 = Brow[gs + 2 + gb];
            v16bf bhi = comb(b0.hi, b1.hi);
            v16bf blo = comb(b0.lo, b1.lo);
            acc[t] = wmma_bf16(ahi, bhi, acc[t]);
            acc[t] = wmma_bf16(ahi, blo, acc[t]);
            acc[t] = wmma_bf16(alo, bhi, acc[t]);
        }
    }
}

// ---------------------------------------------------------------------------
// P0: elementwise split-pack  (f32 array -> bgroup stream), 1 group/thread
// ---------------------------------------------------------------------------
__global__ __launch_bounds__(256) void split_pack_kernel(
    const float* __restrict__ src, uint4* __restrict__ dst) {
    const size_t g = (size_t)blockIdx.x * 256 + threadIdx.x;
    const float4* p = reinterpret_cast<const float4*>(src) + 2 * g;
    float4 x0 = p[0], x1 = p[1];
    float xs[8] = {x0.x, x0.y, x0.z, x0.w, x1.x, x1.y, x1.z, x1.w};
    uint4 hv, lv;
    split8(xs, hv, lv);
    dst[2 * g]     = hv;
    dst[2 * g + 1] = lv;
}

// ---------------------------------------------------------------------------
// P1: transpose + split-pack  s2t_pk[b][d][m-groups] from s2[b][m][d]
//   grid (4, 64, 16), block 256: thread -> (d, m-group); reads coalesced in d.
// ---------------------------------------------------------------------------
__global__ __launch_bounds__(256) void transpose_split_kernel(
    const float* __restrict__ s2, uint4* __restrict__ s2tpk) {
    const int d = blockIdx.x * 256 + threadIdx.x;   // 0..1023
    const int g = blockIdx.y;                       // m-group 0..63
    const int b = blockIdx.z;
    const float* src = s2 + ((size_t)b * 512 + (size_t)g * 8) * 1024 + d;
    float xs[8];
#pragma unroll
    for (int i = 0; i < 8; ++i) xs[i] = src[(size_t)i * 1024];
    uint4 hv, lv;
    split8(xs, hv, lv);
    uint4* row = s2tpk + ((size_t)b * 1024 + d) * 128;  // 64 groups * 2 uint4
    row[2 * g]     = hv;
    row[2 * g + 1] = lv;
}

// ---------------------------------------------------------------------------
// K0: row L2-norms of sent1 (rows 0..8191 -> n1) and sent2 (8192.. -> n2)
// ---------------------------------------------------------------------------
__global__ __launch_bounds__(256) void row_norm_kernel(
    const float* __restrict__ s1, const float* __restrict__ s2,
    float* __restrict__ n1, float* __restrict__ n2) {
    __shared__ float red[256];
    const int row = blockIdx.x;
    const int tid = threadIdx.x;
    const float* src;
    float* dst;
    if (row < 8192) { src = s1 + (size_t)row * 1024; dst = n1 + row; }
    else            { src = s2 + (size_t)(row - 8192) * 1024; dst = n2 + (row - 8192); }
    float sum = 0.f;
#pragma unroll
    for (int j = 0; j < 4; ++j) {
        float v = src[tid + j * 256];
        sum += v * v;
    }
    red[tid] = sum;
    __syncthreads();
    for (int off = 128; off > 0; off >>= 1) {
        if (tid < off) red[tid] += red[tid + off];
        __syncthreads();
    }
    if (tid == 0) *dst = sqrtf(red[0]);
}

// ---------------------------------------------------------------------------
// K2: cos[b] = (S1*S2^T)/max(n1*n2,eps) -> stored PACKED bf16 (GEMM2 A operand)
//   one wave -> 16x64 tile; 4096 waves (512 blocks x 8 waves)
// ---------------------------------------------------------------------------
__global__ __launch_bounds__(256) void gemm1_cos_kernel(
    const bgroup* __restrict__ s1pk, const bgroup* __restrict__ s2pk,
    const float* __restrict__ n1, const float* __restrict__ n2,
    unsigned char* __restrict__ cospk) {
    const int wave = blockIdx.x * 8 + (threadIdx.x >> 5);
    const int lane = threadIdx.x & 31;
    const int b    = wave >> 8;
    const int rem  = wave & 255;
    const int mt   = rem >> 3;            // 0..31
    const int nt   = rem & 7;             // 0..7
    const int ln   = lane & 15;

    const bgroup* Arow = s1pk + ((size_t)b * 512 + (size_t)(mt * 16 + ln)) * 128;
    const bgroup* Bsrc = s2pk + (size_t)b * 512 * 128;

    v8f acc[4] = {};
    gemm_tile_16x64_pk(Arow, Bsrc, nt * 64, 128, lane, acc);

    const int rbase = (lane >= 16) ? 8 : 0;
    float n1v[8];
#pragma unroll
    for (int r = 0; r < 8; ++r)
        n1v[r] = n1[b * 512 + mt * 16 + rbase + r];

#pragma unroll
    for (int t = 0; t < 4; ++t) {
        const int gn  = nt * 64 + t * 16 + ln;
        const int gg  = gn >> 3;
        const int pos = gn & 7;
        const float n2v = n2[b * 512 + gn];
#pragma unroll
        for (int r = 0; r < 8; ++r) {
            const int gm = mt * 16 + rbase + r;
            const float c = acc[t][r] / fmaxf(n1v[r] * n2v, EPSV);
            unsigned short hb = f32_to_bf16_rne(c);
            unsigned short lb = f32_to_bf16_rne(c - bf16_bits_to_f32(hb));
            unsigned char* cr = cospk + ((size_t)b * 512 + gm) * 2048 + gg * 32;
            *(unsigned short*)(cr + pos * 2)      = hb;
            *(unsigned short*)(cr + 16 + pos * 2) = lb;
        }
    }
}

// ---------------------------------------------------------------------------
// K3: sumcos[b,l] = sum_m cos[b,l,m] + 1e-8  (reads packed hi+lo)
// ---------------------------------------------------------------------------
__global__ __launch_bounds__(256) void rowsum_kernel(
    const unsigned char* __restrict__ cospk, float* __restrict__ sums) {
    __shared__ float red[256];
    const int row = blockIdx.x;
    const int tid = threadIdx.x;
    const unsigned char* r = cospk + (size_t)row * 2048;
    const int k0  = tid * 2;                 // two elements per thread
    const int gg  = k0 >> 3;
    const int pos = k0 & 7;                  // even
    unsigned int hu = *(const unsigned int*)(r + gg * 32 + pos * 2);
    unsigned int lu = *(const unsigned int*)(r + gg * 32 + 16 + pos * 2);
    float e0 = __uint_as_float(hu << 16)         + __uint_as_float(lu << 16);
    float e1 = __uint_as_float(hu & 0xFFFF0000u) + __uint_as_float(lu & 0xFFFF0000u);
    red[tid] = e0 + e1;
    __syncthreads();
    for (int off = 128; off > 0; off >>= 1) {
        if (tid < off) red[tid] += red[tid + off];
        __syncthreads();
    }
    if (tid == 0) sums[row] = red[0] + 1e-8f;
}

// ---------------------------------------------------------------------------
// K4: mean[b] = (cos * S2) / sumcos   (A = packed cos, B = packed S2^T)
//   one wave -> 16x64 tile; 8192 waves (1024 blocks x 8 waves)
// ---------------------------------------------------------------------------
__global__ __launch_bounds__(256) void gemm2_mean_kernel(
    const bgroup* __restrict__ cospk, const bgroup* __restrict__ s2tpk,
    const float* __restrict__ sums, float* __restrict__ mean) {
    const int wave = blockIdx.x * 8 + (threadIdx.x >> 5);
    const int lane = threadIdx.x & 31;
    const int b    = wave >> 9;
    const int rem  = wave & 511;
    const int mt   = rem >> 4;            // 0..31
    const int nt   = rem & 15;            // 0..15
    const int ln   = lane & 15;

    const bgroup* Arow = cospk + ((size_t)b * 512 + (size_t)(mt * 16 + ln)) * 64;
    const bgroup* Bsrc = s2tpk + (size_t)b * 1024 * 64;

    v8f acc[4] = {};
    gemm_tile_16x64_pk(Arow, Bsrc, nt * 64, 64, lane, acc);

    const int rbase = (lane >= 16) ? 8 : 0;
    float sv[8];
#pragma unroll
    for (int r = 0; r < 8; ++r)
        sv[r] = sums[b * 512 + mt * 16 + rbase + r];

#pragma unroll
    for (int t = 0; t < 4; ++t) {
        const int gn = nt * 64 + t * 16 + ln;
#pragma unroll
        for (int r = 0; r < 8; ++r) {
            const int gm = mt * 16 + rbase + r;
            mean[((size_t)b * 512 + gm) * 1024 + gn] = acc[t][r] / sv[r];
        }
    }
}

// ---------------------------------------------------------------------------
// K5: out[b,l,p] = (k_p^2 * <s1,mean>) / max(k_p^2 * n1 * ||mean||, eps)
// ---------------------------------------------------------------------------
__global__ __launch_bounds__(256) void final_kernel(
    const float* __restrict__ s1, const float* __restrict__ mean,
    const float* __restrict__ n1, const float* __restrict__ kern,
    float* __restrict__ out) {
    __shared__ float redd[256];
    __shared__ float redn[256];
    const int row = blockIdx.x;
    const int tid = threadIdx.x;
    const float* a = s1 + (size_t)row * 1024;
    const float* m = mean + (size_t)row * 1024;
    float sd = 0.f, sn = 0.f;
#pragma unroll
    for (int j = 0; j < 4; ++j) {
        float x = a[tid + j * 256];
        float y = m[tid + j * 256];
        sd += x * y;
        sn += y * y;
    }
    redd[tid] = sd;
    redn[tid] = sn;
    __syncthreads();
    for (int off = 128; off > 0; off >>= 1) {
        if (tid < off) { redd[tid] += redd[tid + off]; redn[tid] += redn[tid + off]; }
        __syncthreads();
    }
    if (tid < 10) {
        const float dot = redd[0];
        const float nm  = sqrtf(redn[0]);
        const float nn1 = n1[row];
        const float k   = kern[tid];
        const float k2  = k * k;
        out[(size_t)row * 10 + tid] = (k2 * dot) / fmaxf(k2 * nn1 * nm, EPSV);
    }
}

// ---------------------------------------------------------------------------
// Launcher
// ---------------------------------------------------------------------------
extern "C" void kernel_launch(void* const* d_in, const int* in_sizes, int n_in,
                              void* d_out, int out_size, void* d_ws, size_t ws_size,
                              hipStream_t stream) {
    (void)in_sizes; (void)n_in; (void)out_size; (void)ws_size;
    const float* s1   = (const float*)d_in[0];   // [16,512,1024]
    const float* s2   = (const float*)d_in[1];   // [16,512,1024]
    const float* kern = (const float*)d_in[2];   // [10,1]
    float* out = (float*)d_out;                  // [16,512,10]

    // Workspace layout (bytes), total ~144 MiB:
    unsigned char* ws    = (unsigned char*)d_ws;
    unsigned char* s1pk  = ws;                       // packed bf16 sent1: 32 MiB
    unsigned char* s2pk  = s1pk  + 33554432;         // packed bf16 sent2: 32 MiB
    unsigned char* s2tpk = s2pk  + 33554432;         // packed bf16 S2^T : 32 MiB
    unsigned char* cospk = s2tpk + 33554432;         // packed bf16 cos  : 16 MiB
    float* mean = (float*)(cospk + 16777216);        // f32 mean         : 32 MiB
    float* n1   = (float*)((unsigned char*)mean + 33554432);
    float* n2   = n1 + 8192;
    float* sums = n2 + 8192;

    // one-time operand packing (8,388,608 elems each -> 1,048,576 groups)
    split_pack_kernel<<<4096, 256, 0, stream>>>(s1, (uint4*)s1pk);
    split_pack_kernel<<<4096, 256, 0, stream>>>(s2, (uint4*)s2pk);
    transpose_split_kernel<<<dim3(4, 64, 16), 256, 0, stream>>>(s2, (uint4*)s2tpk);
    row_norm_kernel<<<16384, 256, 0, stream>>>(s1, s2, n1, n2);

    gemm1_cos_kernel<<<512, 256, 0, stream>>>((const bgroup*)s1pk, (const bgroup*)s2pk,
                                              n1, n2, cospk);
    rowsum_kernel<<<8192, 256, 0, stream>>>(cospk, sums);
    gemm2_mean_kernel<<<1024, 256, 0, stream>>>((const bgroup*)cospk, (const bgroup*)s2tpk,
                                                sums, mean);
    final_kernel<<<8192, 256, 0, stream>>>(s1, mean, n1, kern, out);
}
